// SocialLSTM_87677462380870
// MI455X (gfx1250) — matrix operand order repaired and checked by reading, not compiled, not run on hardware
//
#include <hip/hip_runtime.h>
#include <hip/hip_bf16.h>
#include <math.h>
#include <stdint.h>

// ---------------- model dims ----------------
#define A_N    512      // agents
#define H_N    128      // hidden
#define E_N    64       // embed
#define G_N    8        // grid side
#define G2_N   64       // grid cells
#define NY_N   8192     // G2*H   (N of Y-GEMM)
#define XK_N   320      // E + H(soc) + H(h)  (K of Z-GEMM)
#define Z_N    512      // 4H     (N of Z-GEMM)
#define TOBS   8
#define PREDN  12
#define NB_F   32.0f
#define CELLW  8.0f     // 2*NB/G

#define KT_Y   4        // K/32 for Y-GEMM (K = 128)
#define KT_Z   10       // K/32 for Z-GEMM (K = 320)

typedef __attribute__((ext_vector_type(16))) _Float16 v16h;
typedef __attribute__((ext_vector_type(8)))  float    v8f;

// ---------------- WMMA fragment swizzle helpers (ISA 7.12.2, 16-bit layouts) ----
__device__ __forceinline__ int afrag_idx(int KT, int m, int k) {
    int mt = m >> 4, kk = k >> 5, kb = k & 31;
    int lhi = (kb >> 3) & 1;
    int e   = (kb & 7) + ((kb >> 4) << 3);
    int l   = (lhi << 4) + (m & 15);
    return (((mt * KT + kk) << 5) + l) * 16 + e;
}
__device__ __forceinline__ int bfrag_idx(int KT, int k, int n) {
    int nt = n >> 4, kk = k >> 5, kb = k & 31;
    int l  = (n & 15) + (((kb >> 3) & 1) << 4);
    int e  = (kb & 7) + ((kb >> 4) << 3);
    return (((nt * KT + kk) << 5) + l) * 16 + e;
}

__device__ __forceinline__ float cleanf(float x) { return (x != x) ? 0.0f : x; }
__device__ __forceinline__ float sigmf(float x)  { return 1.0f / (1.0f + expf(-x)); }

// issue one 16B global->LDS async copy (ASYNCcnt-tracked, ISA 10.x async ops)
__device__ __forceinline__ void async_copy16(const _Float16* gsrc, _Float16* lds_dst) {
    unsigned           lds   = (unsigned)(uintptr_t)lds_dst;   // low 32 bits of the pointer give the LDS address
    unsigned long long gaddr = (unsigned long long)(uintptr_t)gsrc;
    asm volatile("global_load_async_to_lds_b128 %0, %1, off"
                 :: "v"(lds), "v"(gaddr) : "memory");
}
__device__ __forceinline__ void wait_async0() {
    asm volatile("s_wait_asynccnt 0x0" ::: "memory");
}

// ---------------- one-time: zero state ----------------
__global__ void k_init_state(float* __restrict__ h, float* __restrict__ c) {
    int i = blockIdx.x * blockDim.x + threadIdx.x;
    if (i < A_N * H_N) { h[i] = 0.0f; c[i] = 0.0f; }
}

// ---------------- one-time: repack W_p -> f16 B fragments -------------
__global__ void k_cvt_wp(const float* __restrict__ W_p, _Float16* __restrict__ Bp_sw) {
    int idx = blockIdx.x * blockDim.x + threadIdx.x;     // N*K = 8192*128
    int n = idx >> 7;
    int k = idx & 127;
    int g = n >> 7, o = n & 127;
    float v = W_p[o * NY_N + g * H_N + k];
    Bp_sw[bfrag_idx(KT_Y, k, n)] = (_Float16)v;
}

// ---------------- one-time: repack [W_ih | W_hh] -> f16 B fragments ---
__global__ void k_cvt_wz(const float* __restrict__ W_ih, const float* __restrict__ W_hh,
                         _Float16* __restrict__ Bz_sw) {
    int idx = blockIdx.x * blockDim.x + threadIdx.x;     // N*K = 512*320
    if (idx >= Z_N * XK_N) return;
    int n = idx / XK_N;
    int k = idx - n * XK_N;
    float v = (k < E_N + H_N) ? W_ih[n * (E_N + H_N) + k]
                              : W_hh[n * H_N + (k - (E_N + H_N))];
    Bz_sw[bfrag_idx(KT_Z, k, n)] = (_Float16)v;
}

// ---------------- per step: embed + snapshot h into A fragments -------
__global__ void k_embed(const float* __restrict__ pos, const float* __restrict__ W_e,
                        const float* __restrict__ b_e, const float* __restrict__ h,
                        _Float16* __restrict__ hh_sw, _Float16* __restrict__ X_sw) {
    int a = blockIdx.x, t = threadIdx.x;                 // 512 blocks x 128
    float px = cleanf(pos[a * 2 + 0]);
    float py = cleanf(pos[a * 2 + 1]);
    if (t < E_N) {
        float v = W_e[t * 2 + 0] * px + W_e[t * 2 + 1] * py + b_e[t];
        v = v > 0.0f ? v : 0.0f;
        X_sw[afrag_idx(KT_Z, a, t)] = (_Float16)v;
    }
    float hv = h[a * H_N + t];
    hh_sw[afrag_idx(KT_Y, a, t)]            = (_Float16)hv;
    X_sw[afrag_idx(KT_Z, a, E_N + H_N + t)] = (_Float16)hv;
}

// ---------------- per step: Y = hh @ Bp  (M=512,N=8192,K=128), f16 out -
// 256 threads = 8 waves; block owns one mt and 8 consecutive nt tiles.
// A fragments (4KB) staged once via async-to-LDS; B streamed from L2.
__global__ void __launch_bounds__(256) k_ygemm(const _Float16* __restrict__ hh_sw,
                                               const _Float16* __restrict__ Bp_sw,
                                               _Float16* __restrict__ Y) {
    __shared__ _Float16 Ash[KT_Y * 512];                 // 4 kk-blocks x 512 halves
    int tid  = threadIdx.x;
    int lane = tid & 31;
    int wave = tid >> 5;
    int mt = blockIdx.x >> 6;                            // 32 mt
    int nt = ((blockIdx.x & 63) << 3) + wave;            // 512 nt
    async_copy16(hh_sw + (size_t)mt * (KT_Y * 512) + tid * 8, &Ash[tid * 8]);
    wait_async0();
    __syncthreads();
    const v16h* As = (const v16h*)Ash;
    const v16h* Bp = (const v16h*)Bp_sw;
    v8f acc = {};
#pragma unroll
    for (int kk = 0; kk < KT_Y; ++kk) {
        v16h af = As[(kk << 5) + lane];                  // ds_load_b128 x2
        v16h bf = Bp[((nt * KT_Y + kk) << 5) + lane];
        if (kk + 1 < KT_Y)
            __builtin_prefetch(&Bp[((nt * KT_Y + kk + 1) << 5) + lane], 0, 1);
        acc = __builtin_amdgcn_wmma_f32_16x16x32_f16(false, af, false, bf,
                                                     (short)0, acc, false, false);
    }
    int n    = (nt << 4) + (lane & 15);
    int mrow = (mt << 4) + ((lane >> 4) << 3);
#pragma unroll
    for (int r = 0; r < 8; ++r)
        Y[(mrow + r) * NY_N + n] = (_Float16)acc[r];
}

// ---------------- per step: soc[i] = sum_j valid * Y[j, cell(i,j), :] --
__global__ void k_gather(const float* __restrict__ pos, const unsigned char* __restrict__ mask,
                         const _Float16* __restrict__ Y, const float* __restrict__ b_p,
                         _Float16* __restrict__ X_sw) {
    __shared__ int cellLds[A_N];
    int i = blockIdx.x, t = threadIdx.x;                 // 512 blocks x 128
    float pix = cleanf(pos[i * 2 + 0]);
    float piy = cleanf(pos[i * 2 + 1]);
    int mi = mask[i] ? 1 : 0;
    for (int j = t; j < A_N; j += 128) {
        float rx = cleanf(pos[j * 2 + 0]) - pix;
        float ry = cleanf(pos[j * 2 + 1]) - piy;
        int ok = mi && (mask[j] != 0) && (j != i) &&
                 (fabsf(rx) < NB_F) && (fabsf(ry) < NB_F);
        int col = (int)floorf((rx + NB_F) / CELLW);
        col = col < 0 ? 0 : (col > G_N - 1 ? G_N - 1 : col);
        int row = (int)floorf((ry + NB_F) / CELLW);
        row = row < 0 ? 0 : (row > G_N - 1 ? G_N - 1 : row);
        cellLds[j] = ok ? (row * G_N + col) : -1;
    }
    __syncthreads();
    float acc = 0.0f;
    for (int j = 0; j < A_N; ++j) {
        int cc = cellLds[j];
        if (cc >= 0) acc += (float)Y[j * NY_N + cc * H_N + t];
    }
    float soc = mi ? (acc + b_p[t]) : 0.0f;
    X_sw[afrag_idx(KT_Z, i, E_N + t)] = (_Float16)soc;
}

// ---------------- per step: z = X @ Bz  (M=512,N=512,K=320), f32 out ---
// Block owns one mt and 8 nt tiles; A fragments (10KB) staged via async.
__global__ void __launch_bounds__(256) k_zgemm(const _Float16* __restrict__ X_sw,
                                               const _Float16* __restrict__ Bz_sw,
                                               float* __restrict__ z) {
    __shared__ _Float16 Ash[KT_Z * 512];                 // 10 kk-blocks x 512 halves
    int tid  = threadIdx.x;
    int lane = tid & 31;
    int wave = tid >> 5;
    int mt = blockIdx.x >> 2;                            // 32 mt
    int nt = ((blockIdx.x & 3) << 3) + wave;             // 32 nt
    for (int i = tid; i < (KT_Z * 512) / 8; i += 256)    // 640 x 16B chunks
        async_copy16(X_sw + (size_t)mt * (KT_Z * 512) + i * 8, &Ash[i * 8]);
    wait_async0();
    __syncthreads();
    const v16h* As = (const v16h*)Ash;
    const v16h* Bp = (const v16h*)Bz_sw;
    v8f acc = {};
#pragma unroll
    for (int kk = 0; kk < KT_Z; ++kk) {
        v16h af = As[(kk << 5) + lane];
        v16h bf = Bp[((nt * KT_Z + kk) << 5) + lane];
        if (kk + 1 < KT_Z)
            __builtin_prefetch(&Bp[((nt * KT_Z + kk + 1) << 5) + lane], 0, 1);
        acc = __builtin_amdgcn_wmma_f32_16x16x32_f16(false, af, false, bf,
                                                     (short)0, acc, false, false);
    }
    int n    = (nt << 4) + (lane & 15);
    int mrow = (mt << 4) + ((lane >> 4) << 3);
#pragma unroll
    for (int r = 0; r < 8; ++r)
        z[(mrow + r) * Z_N + n] = acc[r];
}

// ---------------- per step: gates, state update, (decoder) output head -
__global__ void k_lstm(const float* __restrict__ z, const float* __restrict__ b_ih,
                       const float* __restrict__ b_hh, float* __restrict__ h,
                       float* __restrict__ c, const float* __restrict__ W_o,
                       const float* __restrict__ b_o, const unsigned char* __restrict__ mask,
                       float* __restrict__ cur, float* __restrict__ out, int dec_step) {
    __shared__ float hsh[H_N];
    __shared__ float raw[5];
    int a = blockIdx.x, t = threadIdx.x;                 // 512 blocks x 128
    const float* za = z + a * Z_N;
    float iv = za[t]           + b_ih[t]           + b_hh[t];
    float fv = za[H_N + t]     + b_ih[H_N + t]     + b_hh[H_N + t];
    float gv = za[2 * H_N + t] + b_ih[2 * H_N + t] + b_hh[2 * H_N + t];
    float ov = za[3 * H_N + t] + b_ih[3 * H_N + t] + b_hh[3 * H_N + t];
    float ig = sigmf(iv), fg = sigmf(fv), og = sigmf(ov);
    float gg = tanhf(gv);
    float cn = fg * c[a * H_N + t] + ig * gg;
    float hn = og * tanhf(cn);
    c[a * H_N + t] = cn;
    h[a * H_N + t] = hn;
    if (dec_step >= 0) {
        hsh[t] = hn;
        __syncthreads();
        if (t < 5) {
            float s = b_o[t];
            for (int k2 = 0; k2 < H_N; ++k2) s += W_o[t * H_N + k2] * hsh[k2];
            raw[t] = s;
        }
        __syncthreads();
        if (t == 0) {
            float m  = mask[a] ? 1.0f : 0.0f;
            float cx = cur[a * 2 + 0] + raw[0] * m;
            float cy = cur[a * 2 + 1] + raw[1] * m;
            cur[a * 2 + 0] = cx;
            cur[a * 2 + 1] = cy;
            float* mu = out;
            float* sg = out + PREDN * A_N * 2;
            float* rh = out + PREDN * A_N * 4;
            mu[(dec_step * A_N + a) * 2 + 0] = cx;
            mu[(dec_step * A_N + a) * 2 + 1] = cy;
            sg[(dec_step * A_N + a) * 2 + 0] = expf(raw[2]) + 1e-6f;
            sg[(dec_step * A_N + a) * 2 + 1] = expf(raw[3]) + 1e-6f;
            rh[dec_step * A_N + a]           = tanhf(raw[4]);
        }
    }
}

// ---------------- cur0 = mask ? obs_clean[-1] : 0 ----------------------
__global__ void k_cur0(const float* __restrict__ obs_last, const unsigned char* __restrict__ mask,
                       float* __restrict__ cur) {
    int idx = blockIdx.x * blockDim.x + threadIdx.x;     // A*2
    if (idx >= A_N * 2) return;
    int a = idx >> 1;
    cur[idx] = mask[a] ? cleanf(obs_last[idx]) : 0.0f;
}

// ======================================================================
extern "C" void kernel_launch(void* const* d_in, const int* in_sizes, int n_in,
                              void* d_out, int out_size, void* d_ws, size_t ws_size,
                              hipStream_t stream) {
    (void)in_sizes; (void)n_in; (void)out_size; (void)ws_size;
    const float*         obs  = (const float*)d_in[0];
    const unsigned char* mask = (const unsigned char*)d_in[1];
    const float* W_e  = (const float*)d_in[2];
    const float* b_e  = (const float*)d_in[3];
    const float* W_p  = (const float*)d_in[4];
    const float* b_p  = (const float*)d_in[5];
    const float* W_ih = (const float*)d_in[6];
    const float* W_hh = (const float*)d_in[7];
    const float* b_ih = (const float*)d_in[8];
    const float* b_hh = (const float*)d_in[9];
    const float* W_o  = (const float*)d_in[10];
    const float* b_o  = (const float*)d_in[11];
    float* out = (float*)d_out;

    char* ws = (char*)d_ws;
    size_t off = 0;
    auto alloc = [&](size_t bytes) -> void* {
        void* p = ws + off;
        off = (off + bytes + 255) & ~(size_t)255;
        return p;
    };
    float*    h     = (float*)alloc((size_t)A_N * H_N * 4);
    float*    c     = (float*)alloc((size_t)A_N * H_N * 4);
    float*    cur   = (float*)alloc((size_t)A_N * 2 * 4);
    float*    z     = (float*)alloc((size_t)A_N * Z_N * 4);
    _Float16* hh_sw = (_Float16*)alloc((size_t)A_N * H_N * 2);
    _Float16* X_sw  = (_Float16*)alloc((size_t)A_N * XK_N * 2);
    _Float16* Y     = (_Float16*)alloc((size_t)A_N * NY_N * 2);
    _Float16* Bp_sw = (_Float16*)alloc((size_t)H_N * NY_N * 2);
    _Float16* Bz_sw = (_Float16*)alloc((size_t)XK_N * Z_N * 2);

    // one-time setup
    k_init_state<<<(A_N * H_N + 255) / 256, 256, 0, stream>>>(h, c);
    k_cvt_wp<<<(NY_N * H_N) / 256, 256, 0, stream>>>(W_p, Bp_sw);
    k_cvt_wz<<<(Z_N * XK_N + 255) / 256, 256, 0, stream>>>(W_ih, W_hh, Bz_sw);

    // encoder scan
    for (int t = 0; t < TOBS; ++t) {
        const float* pos = obs + (size_t)t * A_N * 2;
        k_embed <<<A_N, 128, 0, stream>>>(pos, W_e, b_e, h, hh_sw, X_sw);
        k_ygemm <<<32 * 64, 256, 0, stream>>>(hh_sw, Bp_sw, Y);
        k_gather<<<A_N, 128, 0, stream>>>(pos, mask, Y, b_p, X_sw);
        k_zgemm <<<32 * 4, 256, 0, stream>>>(X_sw, Bz_sw, z);
        k_lstm  <<<A_N, 128, 0, stream>>>(z, b_ih, b_hh, h, c, W_o, b_o,
                                          mask, cur, out, -1);
    }
    k_cur0<<<(A_N * 2 + 255) / 256, 256, 0, stream>>>(obs + (size_t)(TOBS - 1) * A_N * 2,
                                                      mask, cur);
    // decoder scan
    for (int s = 0; s < PREDN; ++s) {
        k_embed <<<A_N, 128, 0, stream>>>(cur, W_e, b_e, h, hh_sw, X_sw);
        k_ygemm <<<32 * 64, 256, 0, stream>>>(hh_sw, Bp_sw, Y);
        k_gather<<<A_N, 128, 0, stream>>>(cur, mask, Y, b_p, X_sw);
        k_zgemm <<<32 * 4, 256, 0, stream>>>(X_sw, Bz_sw, z);
        k_lstm  <<<A_N, 128, 0, stream>>>(z, b_ih, b_hh, h, c, W_o, b_o,
                                          mask, cur, out, s);
    }
}
